// NNUEModel_73547019977247
// MI455X (gfx1250) — matrix operand order repaired and checked by reading, not compiled, not run on hardware
//
#include <hip/hip_runtime.h>

// ---------- types ----------
using f16 = _Float16;
typedef __attribute__((ext_vector_type(16))) _Float16 v16h;
typedef __attribute__((ext_vector_type(8)))  _Float16 f16x8;
typedef __attribute__((ext_vector_type(2)))  _Float16 h2;
typedef __attribute__((ext_vector_type(8)))  float    v8f;
typedef __attribute__((ext_vector_type(2)))  float    f32x2;
typedef __attribute__((ext_vector_type(4)))  int      i32x4;

#define AS1 __attribute__((address_space(1)))
#define AS3 __attribute__((address_space(3)))

#if defined(__has_builtin)
#if __has_builtin(__builtin_amdgcn_global_load_async_to_lds_b128)
#define HAVE_ASYNC_LDS 1
#endif
#endif

// ---------- model constants ----------
#define L1_DIM   3072
#define HALF1    1536            // L1/2
#define D_FT     3080            // L1 + PSQT
#define COUNT    8               // layer stacks
#define L2_DIM   15
#define L3_DIM   32
#define MAXF     32
#define KCHUNKS  96              // 3072 / 32 (K per f16 WMMA)
#define SCALE_C  (127.0f / 128.0f)

__device__ __forceinline__ float clip01(float v) {
    return fminf(fmaxf(v, 0.0f), 1.0f);
}

// =====================================================================
// Kernel 0: pack merged L1 weights (l1_w + tiled l1f_w) into f16 WMMA
// B-fragment order: wpack[((s*96 + kk)*32 + lane)*16 + e], lane%16 <-> N,
// e <-> K = kk*32 + 8*(lane/16) + e%8 + 16*(e/8). Also mb[r] = l1_b[r] + l1f_b[r%16].
// =====================================================================
__global__ __launch_bounds__(256) void k0_pack(
    const float* __restrict__ l1_w, const float* __restrict__ l1_b,
    const float* __restrict__ l1f_w, const float* __restrict__ l1f_b,
    f16* __restrict__ wpack, float* __restrict__ mb)
{
    int idx  = blockIdx.x * 256 + threadIdx.x;   // 0..393215
    int e    = idx & 15;
    int lane = (idx >> 4) & 31;
    int kk   = (idx >> 9) % KCHUNKS;
    int s    = idx / (KCHUNKS * 512);
    int n    = lane & 15;
    int k    = kk * 32 + ((lane >> 4) << 3) + (e & 7) + ((e >> 3) << 4);
    float v  = l1_w[(size_t)(s * 16 + n) * L1_DIM + k] + l1f_w[(size_t)n * L1_DIM + k];
    wpack[idx] = (f16)v;
    if (idx < COUNT * (L2_DIM + 1))
        mb[idx] = l1_b[idx] + l1f_b[idx & 15];
}

// =====================================================================
// Kernel 1: feature transform + clipped pairwise activation -> f16 acts,
// plus per-batch PSQT term. One workgroup (256 thr) per batch item.
// Thread t owns pair indices p = 2t + chunk*512 + sub (chunk<3, sub<2),
// so every gather is a coalesced b64 (float2) load: 6 loads per feature
// row instead of 12 b32, and activation stores become b32 pairs.
// =====================================================================
__global__ __launch_bounds__(256) void k1_ft(
    const float* __restrict__ us, const float* __restrict__ them,
    const int* __restrict__ wi, const float* __restrict__ wv,
    const int* __restrict__ bi, const float* __restrict__ bv,
    const int* __restrict__ pidx,
    const float* __restrict__ ftw, const float* __restrict__ ftb,
    f16* __restrict__ act, float* __restrict__ psqt)
{
    const int b = blockIdx.x;
    const int t = threadIdx.x;

    __shared__ int   siw[MAXF], sib[MAXF];
    __shared__ float svw[MAXF], svb[MAXF];
    if (t < 32)      { siw[t]      = wi[b * MAXF + t];      svw[t]      = wv[b * MAXF + t]; }
    else if (t < 64) { sib[t - 32] = bi[b * MAXF + t - 32]; svb[t - 32] = bv[b * MAXF + t - 32]; }
    __syncthreads();

    // accumulators: [2*chunk+sub]; *l = dims p, *h = dims p+1536
    float awl[6], awh[6], abl[6], abh[6];
    #pragma unroll
    for (int c = 0; c < 3; c++) {
        int p = 2 * t + c * 512;
        f32x2 blo = *(const f32x2*)(ftb + p);
        f32x2 bhi = *(const f32x2*)(ftb + p + HALF1);
        awl[2*c] = blo.x; awl[2*c+1] = blo.y;  awh[2*c] = bhi.x; awh[2*c+1] = bhi.y;
        abl[2*c] = blo.x; abl[2*c+1] = blo.y;  abh[2*c] = bhi.x; abh[2*c+1] = bhi.y;
    }

    for (int m = 0; m < MAXF; m++) {
        const float* row = ftw + (size_t)siw[m] * D_FT;
        if (m + 1 < MAXF)
            __builtin_prefetch(ftw + (size_t)siw[m + 1] * D_FT + 2 * t, 0, 3);
        float v = svw[m];
        #pragma unroll
        for (int c = 0; c < 3; c++) {
            int p = 2 * t + c * 512;
            f32x2 lo = *(const f32x2*)(row + p);
            f32x2 hi = *(const f32x2*)(row + p + HALF1);
            awl[2*c]   = fmaf(lo.x, v, awl[2*c]);
            awl[2*c+1] = fmaf(lo.y, v, awl[2*c+1]);
            awh[2*c]   = fmaf(hi.x, v, awh[2*c]);
            awh[2*c+1] = fmaf(hi.y, v, awh[2*c+1]);
        }
    }
    for (int m = 0; m < MAXF; m++) {
        const float* row = ftw + (size_t)sib[m] * D_FT;
        if (m + 1 < MAXF)
            __builtin_prefetch(ftw + (size_t)sib[m + 1] * D_FT + 2 * t, 0, 3);
        float v = svb[m];
        #pragma unroll
        for (int c = 0; c < 3; c++) {
            int p = 2 * t + c * 512;
            f32x2 lo = *(const f32x2*)(row + p);
            f32x2 hi = *(const f32x2*)(row + p + HALF1);
            abl[2*c]   = fmaf(lo.x, v, abl[2*c]);
            abl[2*c+1] = fmaf(lo.y, v, abl[2*c+1]);
            abh[2*c]   = fmaf(hi.x, v, abh[2*c]);
            abh[2*c+1] = fmaf(hi.y, v, abh[2*c+1]);
        }
    }

    const float u = us[b], tm = them[b];
    f16* ao = act + (size_t)b * L1_DIM;
    #pragma unroll
    for (int c = 0; c < 3; c++) {
        int p = 2 * t + c * 512;
        h2 o1, o2;
        #pragma unroll
        for (int sub = 0; sub < 2; sub++) {
            int i = 2 * c + sub;
            // l0 = clip(us*[w,b] + them*[b,w]); pairwise products
            float flo = clip01(u * awl[i] + tm * abl[i]);
            float fhi = clip01(u * awh[i] + tm * abh[i]);
            float slo = clip01(u * abl[i] + tm * awl[i]);
            float shi = clip01(u * abh[i] + tm * awh[i]);
            if (sub == 0) { o1.x = (f16)(flo * fhi * SCALE_C); o2.x = (f16)(slo * shi * SCALE_C); }
            else          { o1.y = (f16)(flo * fhi * SCALE_C); o2.y = (f16)(slo * shi * SCALE_C); }
        }
        *(h2*)(ao + p)         = o1;
        *(h2*)(ao + p + HALF1) = o2;
    }

    if (t == 0) {
        int pi = pidx[b];
        float wps = ftb[L1_DIM + pi], bps = ftb[L1_DIM + pi];
        for (int m = 0; m < MAXF; m++)
            wps = fmaf(ftw[(size_t)siw[m] * D_FT + L1_DIM + pi], svw[m], wps);
        for (int m = 0; m < MAXF; m++)
            bps = fmaf(ftw[(size_t)sib[m] * D_FT + L1_DIM + pi], svb[m], bps);
        psqt[b] = (wps - bps) * (u - 0.5f);
    }
}

// =====================================================================
// Kernel 2: L1 GEMM via v_wmma_f32_16x16x32_f16 (all 8 stacks, one wave
// per stack, shared LDS A-tile staged with async-to-LDS), then stack
// selection + tiny L2/out layers fused. Grid B/16, block 256.
// =====================================================================
#define AROW 1544   // LDS row stride in f16 (16B-aligned, bank-skewed)

__global__ __launch_bounds__(256) void k2_stack(
    const f16* __restrict__ act, const f16* __restrict__ wpack,
    const float* __restrict__ mb, const float* __restrict__ psqt,
    const int* __restrict__ lsi,
    const float* __restrict__ l2_w, const float* __restrict__ l2_b,
    const float* __restrict__ out_w, const float* __restrict__ out_b,
    float* __restrict__ out)
{
    const int b0   = blockIdx.x * 16;
    const int tid  = threadIdx.x;
    const int wave = tid >> 5;        // stack index 0..7
    const int lane = tid & 31;

    __shared__ f16   s_act[16 * AROW];          // 48.25 KB
    __shared__ float s_l1[COUNT][16][16];       // 8 KB
    __shared__ float s_l2[16][L3_DIM];          // 2 KB

    v8f acc = {};
    const int mrow  = lane & 15;
    const int khalf = (lane >= 16) ? 8 : 0;

    for (int hv = 0; hv < 2; hv++) {            // K in two 1536 halves
        __syncthreads();
#ifdef HAVE_ASYNC_LDS
        // Async DMA: global -> LDS with no VGPR round-trip (ASYNCcnt)
        for (int c = tid; c < 16 * 192; c += 256) {
            int m = c / 192, kc = c % 192;
            f16* gp = const_cast<f16*>(act) + (size_t)(b0 + m) * L1_DIM + hv * HALF1 + kc * 8;
            __builtin_amdgcn_global_load_async_to_lds_b128(
                (AS1 i32x4*)gp, (AS3 i32x4*)&s_act[m * AROW + kc * 8], 0, 0);
        }
#if __has_builtin(__builtin_amdgcn_s_wait_asynccnt)
        __builtin_amdgcn_s_wait_asynccnt(0);
#else
        asm volatile("s_wait_asynccnt 0x0" ::: "memory");
#endif
#else
        for (int c = tid; c < 16 * 192; c += 256) {
            int m = c / 192, kc = c % 192;
            f16x8 v = *(const f16x8*)(act + (size_t)(b0 + m) * L1_DIM + hv * HALF1 + kc * 8);
            *(f16x8*)(&s_act[m * AROW + kc * 8]) = v;
        }
#endif
        __syncthreads();

        for (int kk = 0; kk < 48; kk++) {
            const int kg = hv * 48 + kk;        // global K-chunk 0..95
            // A fragment: lane -> M=lane%16; two contiguous K-runs of 8
            const f16* ap = &s_act[mrow * AROW + kk * 32 + khalf];
            f16x8 a0 = *(const f16x8*)(ap);
            f16x8 a1 = *(const f16x8*)(ap + 16);
            v16h a = __builtin_shufflevector(a0, a1,
                        0,1,2,3,4,5,6,7,8,9,10,11,12,13,14,15);
            // B fragment: pre-packed 32B per lane, fully coalesced
            const f16* bp = wpack + ((((size_t)wave * KCHUNKS + kg) * 32 + lane) << 4);
            v16h bf = *(const v16h*)bp;
            acc = __builtin_amdgcn_wmma_f32_16x16x32_f16(
                      false, a, false, bf, (short)0, acc, false, false);
        }
    }

    // Write C tile (+ merged bias) to LDS: lane -> N=lane%16, vgpr j -> M
    {
        const int n    = lane & 15;
        const int moff = (lane >= 16) ? 8 : 0;
        const float bias = mb[wave * 16 + n];
        #pragma unroll
        for (int j = 0; j < 8; j++)
            s_l1[wave][moff + j][n] = acc[j] + bias;
    }
    __syncthreads();

    // ---- tail: stack selection + L2 (30->32) + output (32->1) ----
    const int m   = tid >> 4;       // batch row in tile
    const int j   = tid & 15;       // handles l2 outputs j and j+16
    const int b   = b0 + m;
    const int sel = lsi[b];

    float l1c[16];
    #pragma unroll
    for (int n = 0; n < 16; n++) l1c[n] = s_l1[sel][m][n];

    float l1x[2 * L2_DIM];
    #pragma unroll
    for (int i = 0; i < L2_DIM; i++) {
        float v = l1c[i];
        l1x[i]          = clip01(v * v * SCALE_C);
        l1x[L2_DIM + i] = clip01(v);
    }

    float a2 = l2_b[sel * L3_DIM + j];
    float c2 = l2_b[sel * L3_DIM + 16 + j];
    const float* w2a = l2_w + (size_t)(sel * L3_DIM + j) * (2 * L2_DIM);
    const float* w2b = l2_w + (size_t)(sel * L3_DIM + 16 + j) * (2 * L2_DIM);
    #pragma unroll
    for (int i = 0; i < 2 * L2_DIM; i++) {
        a2 = fmaf(w2a[i], l1x[i], a2);
        c2 = fmaf(w2b[i], l1x[i], c2);
    }
    s_l2[m][j]      = clip01(a2);
    s_l2[m][16 + j] = clip01(c2);
    __syncthreads();

    if (j == 0) {
        float o = out_b[sel];
        const float* wo = out_w + sel * L3_DIM;
        #pragma unroll
        for (int n = 0; n < L3_DIM; n++) o = fmaf(wo[n], s_l2[m][n], o);
        o += l1c[L2_DIM] + psqt[b];     // + l1x_out + (wps-bps)*(us-0.5)
        out[b] = o;
    }
}

// =====================================================================
extern "C" void kernel_launch(void* const* d_in, const int* in_sizes, int n_in,
                              void* d_out, int out_size, void* d_ws, size_t ws_size,
                              hipStream_t stream)
{
    const float* us    = (const float*)d_in[0];
    const float* them  = (const float*)d_in[1];
    const int*   wi    = (const int*)  d_in[2];
    const float* wv    = (const float*)d_in[3];
    const int*   bi    = (const int*)  d_in[4];
    const float* bv    = (const float*)d_in[5];
    const int*   pidx  = (const int*)  d_in[6];
    const int*   lsi   = (const int*)  d_in[7];
    const float* ftw   = (const float*)d_in[8];
    const float* ftb   = (const float*)d_in[9];
    const float* l1_w  = (const float*)d_in[10];
    const float* l1_b  = (const float*)d_in[11];
    const float* l1f_w = (const float*)d_in[12];
    const float* l1f_b = (const float*)d_in[13];
    const float* l2_w  = (const float*)d_in[14];
    const float* l2_b  = (const float*)d_in[15];
    const float* out_w = (const float*)d_in[16];
    const float* out_b = (const float*)d_in[17];
    float* out = (float*)d_out;

    const int B = in_sizes[0];           // 4096

    char* ws = (char*)d_ws;
    const size_t act_bytes   = (size_t)B * L1_DIM * sizeof(f16);        // 25.2 MB
    const size_t wpack_bytes = (size_t)COUNT * KCHUNKS * 32 * 16 * 2;   // 768 KB
    f16*   act   = (f16*)(ws);
    f16*   wpack = (f16*)(ws + act_bytes);
    float* mb    = (float*)(ws + act_bytes + wpack_bytes);
    float* psqt  = (float*)(ws + act_bytes + wpack_bytes + 512);

    k0_pack<<<(COUNT * KCHUNKS * 512) / 256, 256, 0, stream>>>(
        l1_w, l1_b, l1f_w, l1f_b, wpack, mb);

    k1_ft<<<B, 256, 0, stream>>>(
        us, them, wi, wv, bi, bv, pidx, ftw, ftb, act, psqt);

    k2_stack<<<B / 16, 256, 0, stream>>>(
        act, wpack, mb, psqt, lsi, l2_w, l2_b, out_w, out_b, out);
}
